// Encoder_69630009802955
// MI455X (gfx1250) — matrix operand order repaired and checked by reading, not compiled, not run on hardware
//
#include <hip/hip_runtime.h>
#include <hip/hip_bf16.h>

// ---------------------------------------------------------------------------
// Problem constants (fixed by setup_inputs in the reference)
// ---------------------------------------------------------------------------
#define NN     50000          // codes
#define TT     1024           // visits
#define FC     128            // F = SD*4
#define CAT    192            // F + 2*Dv
#define H1     256            // NHEAD*NHID (gat1 output width)
#define NH1    4              // gat1 heads
#define C1     64             // gat1 per-head dim
#define H2     128            // VDIM (gat2 output width)
#define NH2    1
#define C2     128
#define GDIM   384            // 3*HDIM
#define HD     128            // HDIM
#define NNZE   32768          // T*K nonzeros in H
#define LEAKY  0.2f

typedef __attribute__((ext_vector_type(16))) _Float16 v16h;
typedef __attribute__((ext_vector_type(8)))  float    v8f;

__device__ __forceinline__ float sigf(float x) { return 1.0f / (1.0f + expf(-x)); }

__device__ __forceinline__ void atomicMaxF(float* addr, float val) {
  int* ai = (int*)addr;
  int old = __float_as_int(*addr);
  while (__int_as_float(old) < val) {
    int assumed = old;
    old = atomicCAS(ai, assumed, __float_as_int(val));
    if (old == assumed) break;
  }
}

// ---------------------------------------------------------------------------
// Sparsity extraction: per-row nonzero counts (wave per row, wave32 ballot)
// ---------------------------------------------------------------------------
__global__ void k_rowcount(const float* __restrict__ H, int* __restrict__ rowcnt,
                           int N, int T) {
  int gt = blockIdx.x * blockDim.x + threadIdx.x;
  int row = gt >> 5, lane = gt & 31;
  if (row >= N) return;
  const float* Hr = H + (size_t)row * T;
  int cnt = 0;
  for (int c = 0; c < T; c += 32) {
    int col = c + lane;
    bool p = (col < T) && (Hr[col] != 0.0f);
    unsigned m = (unsigned)__ballot(p);
    cnt += __popc(m);
  }
  if (lane == 0) rowcnt[row] = cnt;
}

// Single-block dual exclusive prefix scan: edge offsets + active-row ranks.
__global__ void k_scan(const int* __restrict__ rowcnt, int* __restrict__ rowstart,
                       int* __restrict__ rankv, int* __restrict__ dTot, int N) {
  __shared__ int s1[1024];
  __shared__ int s2[1024];
  __shared__ int carry[2];
  int tid = threadIdx.x;
  if (tid == 0) { carry[0] = 0; carry[1] = 0; }
  __syncthreads();
  for (int base = 0; base < N; base += 1024) {
    int i = base + tid;
    int x = (i < N) ? rowcnt[i] : 0;
    int m = (x > 0) ? 1 : 0;
    s1[tid] = x; s2[tid] = m;
    __syncthreads();
    for (int ofs = 1; ofs < 1024; ofs <<= 1) {
      int a = 0, b = 0;
      if (tid >= ofs) { a = s1[tid - ofs]; b = s2[tid - ofs]; }
      __syncthreads();
      s1[tid] += a; s2[tid] += b;
      __syncthreads();
    }
    if (i < N) {
      rowstart[i] = carry[0] + s1[tid] - x;   // exclusive
      rankv[i]    = carry[1] + s2[tid] - m;   // cumsum(mask)-1 for active rows
    }
    __syncthreads();
    if (tid == 1023) { carry[0] += s1[1023]; carry[1] += s2[1023]; }
    __syncthreads();
  }
  if (tid == 0) dTot[0] = carry[0];
}

// Ordered edge-list extraction (row-major: matches jnp.nonzero), + visit counts.
__global__ void k_extract(const float* __restrict__ H, const int* __restrict__ rowstart,
                          int* __restrict__ Vl, int* __restrict__ El,
                          int* __restrict__ cntv, int N, int T) {
  int gt = blockIdx.x * blockDim.x + threadIdx.x;
  int row = gt >> 5, lane = gt & 31;
  if (row >= N) return;
  const float* Hr = H + (size_t)row * T;
  int base = rowstart[row];
  for (int c = 0; c < T; c += 32) {
    int col = c + lane;
    bool p = (col < T) && (Hr[col] != 0.0f);
    unsigned m = (unsigned)__ballot(p);
    int ofs = __popc(m & ((1u << lane) - 1u));
    if (p) {
      Vl[base + ofs] = row;
      El[base + ofs] = col;
      atomicAdd(&cntv[col], 1);
    }
    base += __popc(m);
  }
}

// ---------------------------------------------------------------------------
// Xcat = [X_G (hier-embedding gather) | personal_TE]  -> [N, 192]
// ---------------------------------------------------------------------------
__global__ void k_build_xcat(const int* __restrict__ cl,
                             const float* __restrict__ e0, const float* __restrict__ e1,
                             const float* __restrict__ e2, const float* __restrict__ e3,
                             const float* __restrict__ TE,
                             const int* __restrict__ rowcnt, const int* __restrict__ rankv,
                             float* __restrict__ Xcat, int N) {
  long idx = (long)blockIdx.x * blockDim.x + threadIdx.x;
  if (idx >= (long)N * CAT) return;
  int i = (int)(idx / CAT), j = (int)(idx % CAT);
  float v;
  if (j < FC) {
    int l = j >> 5, jj = j & 31;
    int code = cl[i * 4 + l] - 1;
    const float* tbl = (l == 0) ? e0 : (l == 1) ? e1 : (l == 2) ? e2 : e3;
    v = tbl[(size_t)code * 32 + jj];
  } else {
    v = (rowcnt[i] > 0) ? TE[(size_t)rankv[i] * 64 + (j - FC)] : 0.0f;
  }
  Xcat[(size_t)i * CAT + j] = v;
}

// ---------------------------------------------------------------------------
// WMMA GEMM:  C[M,N] = act( A[M,K] @ B + bias )   (f16 inputs, f32 accumulate)
// ACT: 0=none, 1=sigmoid, 2=relu.  TRANSB: B given as W[N,K] row-major (use W^T).
// One wave (32 lanes) per 16x16 tile; M%16==0, K%32==0, N%16==0.
// ---------------------------------------------------------------------------
template<int ACT, int TRANSB>
__global__ void k_gemm_wmma(const float* __restrict__ A, const float* __restrict__ B,
                            const float* __restrict__ bias, float* __restrict__ C,
                            int M, int K, int N) {
  int tn = blockIdx.x, tm = blockIdx.y;
  int lane = threadIdx.x;          // 0..31
  int hi = lane >> 4, lo = lane & 15;
  v8f acc = {};
  for (int kb = 0; kb < K; kb += 32) {
    v16h a, b;
    // A fragment: row = tm*16+lo ; K(e) = 16*hi + e
    const float* Ar = A + (size_t)(tm * 16 + lo) * K + kb + 16 * hi;
#pragma unroll
    for (int e = 0; e < 8; ++e) a[e] = (_Float16)Ar[e];
#pragma unroll
    for (int e = 0; e < 8; ++e) a[8 + e] = (_Float16)Ar[8 + e];
    // B fragment: N = lo ; K(e) = 16*hi + e
    if (TRANSB) {
      const float* Wr = B + (size_t)(tn * 16 + lo) * K + kb + 16 * hi;
#pragma unroll
      for (int e = 0; e < 16; ++e) b[e] = (_Float16)Wr[e];
    } else {
      const float* Bp = B + (size_t)(kb + 16 * hi) * N + tn * 16 + lo;
#pragma unroll
      for (int e = 0; e < 16; ++e) b[e] = (_Float16)Bp[(size_t)e * N];
    }
    acc = __builtin_amdgcn_wmma_f32_16x16x32_f16(false, a, false, b,
                                                 (short)0, acc, false, false);
  }
  int n = tn * 16 + lo;
  float bv = bias ? bias[n] : 0.0f;
#pragma unroll
  for (int v = 0; v < 8; ++v) {
    int m = tm * 16 + v + 8 * hi;
    float val = acc[v] + bv;
    if (ACT == 1) val = sigf(val);
    else if (ACT == 2) val = fmaxf(val, 0.0f);
    C[(size_t)m * N + n] = val;
  }
}

// ---------------------------------------------------------------------------
// GAT scatter/softmax kernels (edge-parallel, atomics; nnz = 32768)
// ---------------------------------------------------------------------------
__global__ void k_fillf(float* __restrict__ p, float v, long n) {
  long i = (long)blockIdx.x * blockDim.x + threadIdx.x;
  if (i < n) p[i] = v;
}

// dst[E[k], :] += src[V[k], :]     (also used for visit_emb = H^T @ X)
__global__ void k_scatter_rows(const int* __restrict__ Vl, const int* __restrict__ El,
                               const int* __restrict__ dTot,
                               const float* __restrict__ src, float* __restrict__ dst,
                               int HC) {
  int k = blockIdx.x;
  if (k >= *dTot) return;
  int v = Vl[k], e = El[k];
  const float* s = src + (size_t)v * HC;
  float* d = dst + (size_t)e * HC;
  for (int j = threadIdx.x; j < HC; j += blockDim.x)
    atomicAdd(&d[j], s[j]);
}

__global__ void k_scale_xe(float* __restrict__ Xe, const int* __restrict__ cntv,
                           int T, int HC) {
  long idx = (long)blockIdx.x * blockDim.x + threadIdx.x;
  if (idx >= (long)T * HC) return;
  int t = (int)(idx / HC);
  Xe[idx] /= fmaxf((float)cntv[t], 1.0f);
}

// ae[t,h] = sum_c Xe[t,h,c] * att[h,c]
__global__ void k_att_score(const float* __restrict__ Xe, const float* __restrict__ att,
                            float* __restrict__ ae, int T, int Hh, int C) {
  int idx = blockIdx.x * blockDim.x + threadIdx.x;
  if (idx >= T * Hh) return;
  int t = idx / Hh, h = idx % Hh;
  const float* xr = Xe + (size_t)t * Hh * C + (size_t)h * C;
  const float* ar = att + (size_t)h * C;
  float s = 0.0f;
  for (int c = 0; c < C; ++c) s += xr[c] * ar[c];
  ae[idx] = s;
}

// a = leaky_relu(ae[E[k],h]); segment max over V
__global__ void k_edge_max(const int* __restrict__ Vl, const int* __restrict__ El,
                           const int* __restrict__ dTot,
                           const float* __restrict__ ae, float* __restrict__ aedge,
                           float* __restrict__ amax, int Hh) {
  long idx = (long)blockIdx.x * blockDim.x + threadIdx.x;
  int k = (int)(idx / Hh);
  if (k >= *dTot) return;
  int h = (int)(idx % Hh);
  float a = ae[El[k] * Hh + h];
  a = (a >= 0.0f) ? a : LEAKY * a;
  aedge[idx] = a;
  atomicMaxF(&amax[Vl[k] * Hh + h], a);
}

// ex = exp(a - amax[V]); den[V] += ex   (ex stored in place of aedge)
__global__ void k_edge_exp(const int* __restrict__ Vl, const int* __restrict__ dTot,
                           float* __restrict__ aedge, const float* __restrict__ amax,
                           float* __restrict__ den, int Hh) {
  long idx = (long)blockIdx.x * blockDim.x + threadIdx.x;
  int k = (int)(idx / Hh);
  if (k >= *dTot) return;
  int h = (int)(idx % Hh);
  float ex = expf(aedge[idx] - amax[Vl[k] * Hh + h]);
  aedge[idx] = ex;
  atomicAdd(&den[Vl[k] * Hh + h], ex);
}

// Xv[V[k], h, c] += Xe[E[k], h, c] * (ex / den[V[k],h])
__global__ void k_edge_scatter_v(const int* __restrict__ Vl, const int* __restrict__ El,
                                 const int* __restrict__ dTot,
                                 const float* __restrict__ exv, const float* __restrict__ den,
                                 const float* __restrict__ Xe, float* __restrict__ Xv,
                                 int Hh, int C) {
  int k = blockIdx.x;
  if (k >= *dTot) return;
  int v = Vl[k], e = El[k];
  int HC = Hh * C;
  const float* s = Xe + (size_t)e * HC;
  float* d = Xv + (size_t)v * HC;
  for (int j = threadIdx.x; j < HC; j += blockDim.x) {
    int h = j / C;
    float alpha = exv[k * Hh + h] / den[v * Hh + h];
    atomicAdd(&d[j], s[j] * alpha);
  }
}

// B = act(A + B) elementwise (residual + optional ReLU), in place over B
__global__ void k_combine(const float* __restrict__ A, float* __restrict__ B,
                          int relu, long n) {
  long i = (long)blockIdx.x * blockDim.x + threadIdx.x;
  if (i >= n) return;
  float x = A[i] + B[i];
  if (relu) x = fmaxf(x, 0.0f);
  B[i] = x;
}

// ---------------------------------------------------------------------------
// PGatt blend: X = a0*log_softmax(X2) + (1-a0)*X_G, a0 from sigmoid-proj gates
// One block (128 threads) per row. X_G = first 128 cols of Xcat (stride 192).
// ---------------------------------------------------------------------------
__global__ void k_blend(float* __restrict__ X2, const float* __restrict__ Xcat,
                        const float* __restrict__ WF, const float* __restrict__ zw,
                        int N) {
  __shared__ float xp[128];
  __shared__ float xg[128];
  __shared__ float red[128];
  int i = blockIdx.x, tid = threadIdx.x;
  float x = X2[(size_t)i * FC + tid];
  xg[tid] = Xcat[(size_t)i * CAT + tid];
  red[tid] = x;
  __syncthreads();
  for (int o = 64; o > 0; o >>= 1) { if (tid < o) red[tid] = fmaxf(red[tid], red[tid + o]); __syncthreads(); }
  float m = red[0];
  __syncthreads();
  red[tid] = expf(x - m);
  __syncthreads();
  for (int o = 64; o > 0; o >>= 1) { if (tid < o) red[tid] += red[tid + o]; __syncthreads(); }
  float lse = m + logf(red[0]);
  __syncthreads();
  xp[tid] = x - lse;
  __syncthreads();
  float sP = 0.0f, sG = 0.0f;
  if (tid < 64) {
    float dP = 0.0f, dG = 0.0f;
    for (int j = 0; j < 128; ++j) {
      float w = WF[j * 64 + tid];
      dP += xp[j] * w;
      dG += xg[j] * w;
    }
    float z = zw[tid];
    sP = z * sigf(dP);
    sG = z * sigf(dG);
  }
  red[tid] = (tid < 64) ? sP : 0.0f;
  __syncthreads();
  for (int o = 64; o > 0; o >>= 1) { if (tid < o) red[tid] += red[tid + o]; __syncthreads(); }
  float SP = red[0];
  __syncthreads();
  red[tid] = (tid < 64) ? sG : 0.0f;
  __syncthreads();
  for (int o = 64; o > 0; o >>= 1) { if (tid < o) red[tid] += red[tid + o]; __syncthreads(); }
  float SG = red[0];
  __syncthreads();
  float zP = expf(SP), zG = expf(SG);
  float a0 = zP / (zP + zG);
  X2[(size_t)i * FC + tid] = a0 * xp[tid] + (1.0f - a0) * xg[tid];
}

// ---------------------------------------------------------------------------
// GRU scan: GI[t] = W_ih @ x_t + b_ih (precomputed via WMMA); here per step
// gh = W_hh @ h + b_hh; gates; write hs. Single block, 384 threads.
// ---------------------------------------------------------------------------
__global__ void k_gru(const float* __restrict__ GI, const float* __restrict__ Whh,
                      const float* __restrict__ bhh, float* __restrict__ hs, int T) {
  __shared__ float h[HD];
  __shared__ float gh[GDIM];
  int tid = threadIdx.x;    // 0..383
  if (tid < HD) h[tid] = 0.0f;
  __syncthreads();
  for (int t = 0; t < T; ++t) {
    float acc = bhh[tid];
    const float* wr = Whh + (size_t)tid * HD;
#pragma unroll 4
    for (int k = 0; k < HD; ++k) acc += wr[k] * h[k];
    gh[tid] = acc;
    __syncthreads();
    float hnew = 0.0f;
    if (tid < HD) {
      const float* gi = GI + (size_t)t * GDIM;
      float r = sigf(gi[tid] + gh[tid]);
      float z = sigf(gi[HD + tid] + gh[HD + tid]);
      float n = tanhf(gi[2 * HD + tid] + r * gh[2 * HD + tid]);
      hnew = (1.0f - z) * n + z * h[tid];
    }
    __syncthreads();
    if (tid < HD) { h[tid] = hnew; hs[(size_t)t * HD + tid] = hnew; }
    __syncthreads();
  }
}

// Final temporal attention: out = sum_t softmax(hs @ w)[t] * hs[t]
__global__ void k_attn(const float* __restrict__ hs, const float* __restrict__ w,
                       float* __restrict__ out, int T) {
  __shared__ float sc[TT];
  __shared__ float red[256];
  int tid = threadIdx.x;    // 256
  for (int t = tid; t < T; t += 256) {
    const float* hr = hs + (size_t)t * HD;
    float s = 0.0f;
    for (int j = 0; j < HD; ++j) s += hr[j] * w[j];
    sc[t] = s;
  }
  __syncthreads();
  float m = -1e30f;
  for (int t = tid; t < T; t += 256) m = fmaxf(m, sc[t]);
  red[tid] = m;
  __syncthreads();
  for (int o = 128; o > 0; o >>= 1) { if (tid < o) red[tid] = fmaxf(red[tid], red[tid + o]); __syncthreads(); }
  m = red[0];
  __syncthreads();
  float sum = 0.0f;
  for (int t = tid; t < T; t += 256) { float e = expf(sc[t] - m); sc[t] = e; sum += e; }
  red[tid] = sum;
  __syncthreads();
  for (int o = 128; o > 0; o >>= 1) { if (tid < o) red[tid] += red[tid + o]; __syncthreads(); }
  float den = red[0];
  __syncthreads();
  if (tid < HD) {
    float acc = 0.0f;
    for (int t = 0; t < T; ++t) acc += sc[t] * hs[(size_t)t * HD + tid];
    out[tid] = acc / den;
  }
}

// ---------------------------------------------------------------------------
extern "C" void kernel_launch(void* const* d_in, const int* in_sizes, int n_in,
                              void* d_out, int out_size, void* d_ws, size_t ws_size,
                              hipStream_t stream) {
  (void)in_sizes; (void)n_in; (void)out_size; (void)ws_size;
  const float* Hm   = (const float*)d_in[0];
  const float* TE   = (const float*)d_in[1];
  const int*   clv  = (const int*)  d_in[2];
  const float* e0   = (const float*)d_in[3];
  const float* e1   = (const float*)d_in[4];
  const float* e2   = (const float*)d_in[5];
  const float* e3   = (const float*)d_in[6];
  const float* Wt   = (const float*)d_in[7];
  const float* Wtb  = (const float*)d_in[8];
  const float* WF   = (const float*)d_in[9];
  const float* zw   = (const float*)d_in[10];
  const float* g1W  = (const float*)d_in[11];
  const float* g1a  = (const float*)d_in[12];
  const float* g2W  = (const float*)d_in[13];
  const float* g2a  = (const float*)d_in[14];
  const float* Wih  = (const float*)d_in[15];
  const float* Whh  = (const float*)d_in[16];
  const float* bih  = (const float*)d_in[17];
  const float* bhh  = (const float*)d_in[18];
  const float* actx = (const float*)d_in[19];
  float* out = (float*)d_out;

  // ---- workspace bump allocator -------------------------------------------
  char* wsp = (char*)d_ws;
  size_t off = 0;
  auto alloc = [&](size_t bytes) -> void* {
    void* p = wsp + off;
    off += (bytes + 255) & ~(size_t)255;
    return p;
  };
  int*   rowcnt   = (int*)alloc((size_t)NN * 4);
  int*   rowstart = (int*)alloc((size_t)NN * 4);
  int*   rankv    = (int*)alloc((size_t)NN * 4);
  int*   dTot     = (int*)alloc(256);
  int*   Vl       = (int*)alloc((size_t)NNZE * 4);
  int*   El       = (int*)alloc((size_t)NNZE * 4);
  int*   cntv     = (int*)alloc((size_t)TT * 4);
  float* ae1      = (float*)alloc((size_t)TT * NH1 * 4);
  float* ae2      = (float*)alloc((size_t)TT * 4);
  float* amax1    = (float*)alloc((size_t)NN * NH1 * 4);
  float* den1     = (float*)alloc((size_t)NN * NH1 * 4);
  float* aedge1   = (float*)alloc((size_t)NNZE * NH1 * 4);
  float* amax2    = (float*)alloc((size_t)NN * 4);
  float* den2     = (float*)alloc((size_t)NN * 4);
  float* aedge2   = (float*)alloc((size_t)NNZE * 4);
  float* Xe       = (float*)alloc((size_t)TT * H1 * 4);       // reused for gat2 (T*128)
  float* ve       = (float*)alloc((size_t)TT * HD * 4);
  float* GI       = (float*)alloc((size_t)TT * GDIM * 4);
  float* hsb      = (float*)alloc((size_t)TT * HD * 4);
  float* Xcat     = (float*)alloc((size_t)NN * CAT * 4);      // cols 0..127 == X_G
  float* X0       = (float*)alloc((size_t)NN * FC * 4);       // X_0, later X2lin
  float* X1       = (float*)alloc((size_t)NN * H1 * 4);       // X1lin -> Xg1 (in place)
  float* Xv       = (float*)alloc((size_t)NN * H1 * 4);       // Xv1; later Xv2/Xfinal [N,128]

  // ---- 1) sparsity extraction ---------------------------------------------
  hipMemsetAsync(cntv, 0, (size_t)TT * 4, stream);
  {
    int threads = 256, blocks = (NN * 32 + threads - 1) / threads;
    k_rowcount<<<blocks, threads, 0, stream>>>(Hm, rowcnt, NN, TT);
    k_scan<<<1, 1024, 0, stream>>>(rowcnt, rowstart, rankv, dTot, NN);
    k_extract<<<blocks, threads, 0, stream>>>(Hm, rowstart, Vl, El, cntv, NN, TT);
  }

  // ---- 2) Xcat = [X_G | personal_TE] --------------------------------------
  k_build_xcat<<<((long)NN * CAT + 255) / 256, 256, 0, stream>>>(
      clv, e0, e1, e2, e3, TE, rowcnt, rankv, Xcat, NN);

  // ---- 3) X_0 = sigmoid(Xcat @ W_t + b) -----------------------------------
  k_gemm_wmma<1, 0><<<dim3(FC / 16, NN / 16), 32, 0, stream>>>(Xcat, Wt, Wtb, X0, NN, CAT, FC);

  // ---- 4) GAT layer 1 ------------------------------------------------------
  k_gemm_wmma<0, 0><<<dim3(H1 / 16, NN / 16), 32, 0, stream>>>(X0, g1W, nullptr, X1, NN, FC, H1);
  hipMemsetAsync(Xe, 0, (size_t)TT * H1 * 4, stream);
  hipMemsetAsync(Xv, 0, (size_t)NN * H1 * 4, stream);
  hipMemsetAsync(den1, 0, (size_t)NN * NH1 * 4, stream);
  k_fillf<<<((long)NN * NH1 + 255) / 256, 256, 0, stream>>>(amax1, -1e30f, (long)NN * NH1);
  k_scatter_rows<<<NNZE, 256, 0, stream>>>(Vl, El, dTot, X1, Xe, H1);
  k_scale_xe<<<((long)TT * H1 + 255) / 256, 256, 0, stream>>>(Xe, cntv, TT, H1);
  k_att_score<<<(TT * NH1 + 63) / 64, 64, 0, stream>>>(Xe, g1a, ae1, TT, NH1, C1);
  k_edge_max<<<((long)NNZE * NH1 + 255) / 256, 256, 0, stream>>>(Vl, El, dTot, ae1, aedge1, amax1, NH1);
  k_edge_exp<<<((long)NNZE * NH1 + 255) / 256, 256, 0, stream>>>(Vl, dTot, aedge1, amax1, den1, NH1);
  k_edge_scatter_v<<<NNZE, 256, 0, stream>>>(Vl, El, dTot, aedge1, den1, Xe, Xv, NH1, C1);
  k_combine<<<((long)NN * H1 + 255) / 256, 256, 0, stream>>>(Xv, X1, 1, (long)NN * H1); // X1 = relu(Xv+X1)

  // ---- 5) GAT layer 2 ------------------------------------------------------
  k_gemm_wmma<0, 0><<<dim3(FC / 16, NN / 16), 32, 0, stream>>>(X1, g2W, nullptr, X0, NN, H1, FC); // X2lin -> X0
  hipMemsetAsync(Xe, 0, (size_t)TT * H2 * 4, stream);
  hipMemsetAsync(Xv, 0, (size_t)NN * H2 * 4, stream);  // Xv2
  hipMemsetAsync(den2, 0, (size_t)NN * 4, stream);
  k_fillf<<<((long)NN + 255) / 256, 256, 0, stream>>>(amax2, -1e30f, (long)NN);
  k_scatter_rows<<<NNZE, 128, 0, stream>>>(Vl, El, dTot, X0, Xe, H2);
  k_scale_xe<<<((long)TT * H2 + 255) / 256, 256, 0, stream>>>(Xe, cntv, TT, H2);
  k_att_score<<<(TT + 63) / 64, 64, 0, stream>>>(Xe, g2a, ae2, TT, NH2, C2);
  k_edge_max<<<((long)NNZE + 255) / 256, 256, 0, stream>>>(Vl, El, dTot, ae2, aedge2, amax2, NH2);
  k_edge_exp<<<((long)NNZE + 255) / 256, 256, 0, stream>>>(Vl, dTot, aedge2, amax2, den2, NH2);
  k_edge_scatter_v<<<NNZE, 128, 0, stream>>>(Vl, El, dTot, aedge2, den2, Xe, Xv, NH2, C2);
  k_combine<<<((long)NN * H2 + 255) / 256, 256, 0, stream>>>(X0, Xv, 0, (long)NN * H2); // Xv = Xv + X2lin

  // ---- 6) PGatt blend (in place over Xv -> X_final) ------------------------
  k_blend<<<NN, 128, 0, stream>>>(Xv, Xcat, WF, zw, NN);

  // ---- 7) visit_emb = H^T @ X ---------------------------------------------
  hipMemsetAsync(ve, 0, (size_t)TT * HD * 4, stream);
  k_scatter_rows<<<NNZE, 128, 0, stream>>>(Vl, El, dTot, Xv, ve, HD);

  // ---- 8) GI = visit_emb @ W_ih^T + b_ih  (WMMA, transposed B) -------------
  k_gemm_wmma<0, 1><<<dim3(GDIM / 16, TT / 16), 32, 0, stream>>>(ve, Wih, bih, GI, TT, HD, GDIM);

  // ---- 9) GRU scan + temporal attention -----------------------------------
  k_gru<<<1, GDIM, 0, stream>>>(GI, Whh, bhh, hsb, TT);
  k_attn<<<1, 256, 0, stream>>>(hsb, actx, out, TT);
}
